// BaseMessagePassingLayer_61263413510717
// MI455X (gfx1250) — compile-verified, matrix operand order checked by reading, not run on hardware
//
#include <hip/hip_runtime.h>
#include <hip/hip_bf16.h>
#include <stdint.h>

#define H        128
#define INV      64
#define KDIM     320            // 2H + INV
#define EPB      128            // edges per block
#define TPB      256            // 8 waves
#define KSTEPS   10             // 320 / 32
#define NTILES   8              // 128 / 16
#define LDS_STRIDE 328          // bf16 elems per row; 328*2 = 656 B (16B aligned)

typedef __bf16 bf16_t;
typedef bf16_t v8bf  __attribute__((ext_vector_type(8)));
typedef bf16_t v16bf __attribute__((ext_vector_type(16)));
typedef float  v8f   __attribute__((ext_vector_type(8)));

struct __align__(16) F4 { float x, y, z, w; };

__device__ __forceinline__ unsigned short f2bf(float f) {
    unsigned int u = __float_as_uint(f);
    unsigned int r = u + 0x7FFFu + ((u >> 16) & 1u);   // round-to-nearest-even
    return (unsigned short)(r >> 16);
}

__global__ void zero_out_kernel(float* __restrict__ out, int n) {
    int i = blockIdx.x * blockDim.x + threadIdx.x;
    if (i < n) out[i] = 0.0f;
}

// W1 [K=320][N=128] f32 -> W1T [N=128][K=320] bf16 (N-major, A-style fragment loads)
__global__ void w1_transpose_bf16_kernel(const float* __restrict__ W1,
                                         unsigned short* __restrict__ W1T) {
    int t = blockIdx.x * blockDim.x + threadIdx.x;
    if (t >= KDIM * H) return;
    int n = t / KDIM;
    int k = t - n * KDIM;
    W1T[n * KDIM + k] = f2bf(W1[k * H + n]);
}

__global__ void __launch_bounds__(TPB)
msg_pass_kernel(const float* __restrict__ x_send,
                const float* __restrict__ x_rec,
                const float* __restrict__ edge_attr,
                const int*   __restrict__ index_send,
                const int*   __restrict__ index_rec,
                const unsigned short* __restrict__ W1T,   // [H][KDIM] bf16
                const float* __restrict__ b1,
                const float* __restrict__ Wg,
                const float* __restrict__ bg,
                float* __restrict__ out,
                int E)
{
    extern __shared__ char smem[];
    unsigned short* sW1 = (unsigned short*)smem;           // [H][LDS_STRIDE]
    unsigned short* sA  = sW1 + H * LDS_STRIDE;            // [EPB][LDS_STRIDE]
    int* sIdxSend       = (int*)(sA + EPB * LDS_STRIDE);   // [EPB]
    int* sIdxRec        = sIdxSend + EPB;                  // [EPB]

    const int tid = threadIdx.x;
    const int e0  = blockIdx.x * EPB;

    // ---- stage W1T into LDS via async global->LDS DMA (bypasses VGPRs) ----
    // 128 rows x 40 x 16B chunks; GV mode: per-lane 64-bit global addr,
    // per-lane 32-bit LDS offset (low 32 bits of the flat LDS pointer).
    {
        const uint32_t ldsBase = (uint32_t)(size_t)(void*)sW1;
        const uint64_t gBase   = (uint64_t)(size_t)(const void*)W1T;
        for (int c = tid; c < H * 40; c += TPB) {
            int row = c / 40;
            int col = c - row * 40;                  // 16B chunk within row
            uint32_t lds = ldsBase + (uint32_t)(row * (LDS_STRIDE * 2) + col * 16);
            uint64_t ga  = gBase + (uint64_t)c * 16u;  // contiguous source
            asm volatile("global_load_async_to_lds_b128 %0, %1, off"
                         :: "v"(lds), "v"(ga) : "memory");
        }
    }

    // ---- stage edge indices (overlaps with async DMA above) ----
    if (tid < EPB) {
        int e = e0 + tid;
        sIdxSend[tid] = (e < E) ? index_send[e] : 0;
    } else {
        int e = e0 + (tid - EPB);
        sIdxRec[tid - EPB] = (e < E) ? index_rec[e] : 0;
    }
    __syncthreads();

    // ---- gather + f32->bf16 convert the A tile: 128 edges x 80 float4 chunks ----
    for (int c = tid; c < EPB * 80; c += TPB) {
        int er  = c / 80;                 // edge row within tile
        int q   = c - er * 80;            // float4 chunk index
        int col = q * 4;                  // feature column
        int e   = e0 + er;
        F4 v = {0.f, 0.f, 0.f, 0.f};
        if (e < E) {
            if (col < H) {
                int is = sIdxSend[er];
                v = *(const F4*)(x_send + (size_t)is * H + col);
            } else if (col < 2 * H) {
                int ir = sIdxRec[er];
                v = *(const F4*)(x_rec + (size_t)ir * H + (col - H));
            } else {
                v = *(const F4*)(edge_attr + (size_t)e * INV + (col - 2 * H));
            }
        }
        unsigned int p0 = (unsigned int)f2bf(v.x) | ((unsigned int)f2bf(v.y) << 16);
        unsigned int p1 = (unsigned int)f2bf(v.z) | ((unsigned int)f2bf(v.w) << 16);
        unsigned int* dst = (unsigned int*)(sA + er * LDS_STRIDE + col);
        dst[0] = p0;
        dst[1] = p1;
    }

    // drain this wave's async DMAs into LDS, then block-wide barrier
    asm volatile("s_wait_asynccnt 0" ::: "memory");
    __syncthreads();

    // ---- WMMA GEMM: each wave = one 16-edge M-tile x all 8 N-tiles ----
    const int wave  = tid >> 5;
    const int lane  = tid & 31;
    const int lhalf = lane >> 4;       // 0: K+0..7/16..23 , 1: K+8..15/24..31
    const int l16   = lane & 15;

    v8f acc[NTILES];
    #pragma unroll
    for (int t = 0; t < NTILES; ++t)
        acc[t] = (v8f){0.f, 0.f, 0.f, 0.f, 0.f, 0.f, 0.f, 0.f};

    const unsigned short* aRow = sA + (wave * 16 + l16) * LDS_STRIDE;

    for (int ks = 0; ks < KSTEPS; ++ks) {
        int kb = ks * 32 + lhalf * 8;
        v8bf alo = *(const v8bf*)(aRow + kb);
        v8bf ahi = *(const v8bf*)(aRow + kb + 16);
        v16bf a = __builtin_shufflevector(alo, ahi,
                                          0,1,2,3,4,5,6,7,8,9,10,11,12,13,14,15);
        #pragma unroll
        for (int t = 0; t < NTILES; ++t) {
            const unsigned short* bRow = sW1 + (t * 16 + l16) * LDS_STRIDE;
            v8bf blo = *(const v8bf*)(bRow + kb);
            v8bf bhi = *(const v8bf*)(bRow + kb + 16);
            v16bf b = __builtin_shufflevector(blo, bhi,
                                              0,1,2,3,4,5,6,7,8,9,10,11,12,13,14,15);
            acc[t] = __builtin_amdgcn_wmma_f32_16x16x32_bf16(
                         false, a, false, b, (short)0, acc[t], false, false);
        }
    }

    // ---- epilogue: bias + SiLU, sigmoid gate, scale, atomic scatter ----
    float bias[NTILES], wgv[NTILES];
    #pragma unroll
    for (int t = 0; t < NTILES; ++t) {
        bias[t] = b1[t * 16 + l16];
        wgv[t]  = Wg[t * 16 + l16];
    }
    const float bgv = bg[0];

    float gate[8];
    #pragma unroll
    for (int r = 0; r < 8; ++r) {
        float part = 0.f;
        #pragma unroll
        for (int t = 0; t < NTILES; ++t) {
            float x = acc[t][r] + bias[t];
            float m = x / (1.f + __expf(-x));      // SiLU
            acc[t][r] = m;
            part += m * wgv[t];
        }
        // reduce across the 16 lanes of this half-group (row M lives there)
        part += __shfl_xor(part, 1, 32);
        part += __shfl_xor(part, 2, 32);
        part += __shfl_xor(part, 4, 32);
        part += __shfl_xor(part, 8, 32);
        gate[r] = 1.f / (1.f + __expf(-(part + bgv)));
    }

    #pragma unroll
    for (int r = 0; r < 8; ++r) {
        int m  = r + lhalf * 8;            // row within 16-edge tile (C layout)
        int er = wave * 16 + m;
        int e  = e0 + er;
        if (e >= E) continue;
        int node = sIdxRec[er];
        float* dst = out + (size_t)node * H + l16;
        #pragma unroll
        for (int t = 0; t < NTILES; ++t) {
            __hip_atomic_fetch_add(dst + t * 16, acc[t][r] * gate[r],
                                   __ATOMIC_RELAXED, __HIP_MEMORY_SCOPE_AGENT);
        }
    }
}

extern "C" void kernel_launch(void* const* d_in, const int* in_sizes, int n_in,
                              void* d_out, int out_size, void* d_ws, size_t ws_size,
                              hipStream_t stream) {
    const float* x_send    = (const float*)d_in[0];
    const float* x_rec     = (const float*)d_in[1];
    const float* edge_attr = (const float*)d_in[2];
    const int*   index_send= (const int*)d_in[3];
    const int*   index_rec = (const int*)d_in[4];
    const float* W1        = (const float*)d_in[5];
    const float* b1        = (const float*)d_in[6];
    const float* Wg        = (const float*)d_in[7];
    const float* bg        = (const float*)d_in[8];
    float* out = (float*)d_out;
    const int E = in_sizes[3];

    unsigned short* W1T = (unsigned short*)d_ws;   // 320*128 bf16 = 80 KB

    // 1) zero the output (harness poisons it; we accumulate with atomics)
    zero_out_kernel<<<(out_size + TPB - 1) / TPB, TPB, 0, stream>>>(out, out_size);

    // 2) W1 -> transposed bf16 (stays resident in L2)
    w1_transpose_bf16_kernel<<<(KDIM * H + TPB - 1) / TPB, TPB, 0, stream>>>(W1, W1T);

    // 3) fused gather + GEMM + SiLU + gate + scatter
    int nblocks = (E + EPB - 1) / EPB;
    size_t shmem = (size_t)(H + EPB) * LDS_STRIDE * sizeof(unsigned short)
                 + (size_t)2 * EPB * sizeof(int);          // 168,960 B
    msg_pass_kernel<<<nblocks, TPB, shmem, stream>>>(
        x_send, x_rec, edge_attr, index_send, index_rec,
        W1T, b1, Wg, bg, out, E);
}
